// STNetSR_39917426049706
// MI455X (gfx1250) — matrix operand order repaired and checked by reading, not compiled, not run on hardware
//
#include <hip/hip_runtime.h>
#include <hip/hip_bf16.h>

typedef __attribute__((ext_vector_type(16))) __bf16 v16bf;
typedef __attribute__((ext_vector_type(8)))  __bf16 v8bf;
typedef __attribute__((ext_vector_type(8)))  float  v8f;

#define DIMC 96
#define NSIDE 256
#define NTOK (NSIDE*NSIDE)
#define WS 8
#define OWIN 12
#define HEADS 6
#define HD 16
#define TBL 361
#define NQ 64
#define NK 144
#define NKP 160   // NK padded to multiple of 32

__device__ __forceinline__ v8f wmma_bf16(v16bf a, v16bf b, v8f c) {
    return __builtin_amdgcn_wmma_f32_16x16x32_bf16(false, a, false, b,
                                                   (short)0, c, false, false);
}
__device__ __forceinline__ v8bf zero8() {
    v8bf z;
#pragma unroll
    for (int j = 0; j < 8; ++j) z[j] = (__bf16)0.f;
    return z;
}
__device__ __forceinline__ v16bf mk16(v8bf lo, v8bf hi8) {
    v16bf a;
#pragma unroll
    for (int j = 0; j < 8; ++j) { a[j] = lo[j]; a[j + 8] = hi8[j]; }
    return a;
}
// A fragment: rows of [m][k] memory, contiguous chunks at k0+hi*8 and k0+16+hi*8
__device__ __forceinline__ v16bf frag_a(const __bf16* row_k0, int hi) {
    return mk16(*(const v8bf*)(row_k0 + hi * 8),
                *(const v8bf*)(row_k0 + 16 + hi * 8));
}
// B fragment: rows of [n][k] memory, contiguous chunks at k0+hi*8 and k0+16+hi*8
__device__ __forceinline__ v16bf frag_b(const __bf16* row_k0, int hi) {
    return mk16(*(const v8bf*)(row_k0 + hi * 8),
                *(const v8bf*)(row_k0 + 16 + hi * 8));
}

// ---------------------------------------------------------------------------
// K0: relative position bias table  bias[h][q][k] = rpb[RPI[q][k]][h]
// ---------------------------------------------------------------------------
__global__ void k_bias(const float* __restrict__ rpb, float* __restrict__ bias) {
    int idx = blockIdx.x * blockDim.x + threadIdx.x;
    if (idx >= HEADS * NQ * NK) return;
    int hh  = idx / (NQ * NK);
    int rem = idx % (NQ * NK);
    int q = rem / NK, k = rem % NK;
    int co0 = q >> 3, co1 = q & 7;
    int ce0 = k / OWIN, ce1 = k % OWIN;
    int v = (ce0 - co0 - 3) * (WS + OWIN - 1) + (ce1 - co1 - 3);
    v = ((v % TBL) + TBL) % TBL;
    bias[idx] = rpb[v * HEADS + hh];
}

// ---------------------------------------------------------------------------
// K1: per-token LayerNorm (over 96 channels) -> bf16, one wave per token
// ---------------------------------------------------------------------------
__global__ __launch_bounds__(256) void k_ln(const float* __restrict__ x,
                                            const float* __restrict__ w,
                                            const float* __restrict__ b,
                                            __bf16* __restrict__ y) {
    int wave = threadIdx.x >> 5, lane = threadIdx.x & 31;
    size_t t = (size_t)blockIdx.x * 8 + wave;
    const float* row = x + t * DIMC;
    float v0 = row[lane], v1 = row[lane + 32], v2 = row[lane + 64];
    float s = v0 + v1 + v2, s2 = v0 * v0 + v1 * v1 + v2 * v2;
    for (int off = 16; off; off >>= 1) {
        s  += __shfl_xor(s,  off, 32);
        s2 += __shfl_xor(s2, off, 32);
    }
    float mu  = s  * (1.f / 96.f);
    float var = s2 * (1.f / 96.f) - mu * mu;
    float r = rsqrtf(var + 1e-5f);
    __bf16* yr = y + t * DIMC;
    yr[lane]      = (__bf16)((v0 - mu) * r * w[lane]      + b[lane]);
    yr[lane + 32] = (__bf16)((v1 - mu) * r * w[lane + 32] + b[lane + 32]);
    yr[lane + 64] = (__bf16)((v2 - mu) * r * w[lane + 64] + b[lane + 64]);
}

// ---------------------------------------------------------------------------
// K2: windowed overlapping cross-attention (stage 1). One block per
// (window, direction). 8 waves; each wave owns (head, q-tile) items.
// QK^T computed transposed (D[p][q] = K Q^T) so logit stores and bias loads
// are 8 consecutive values; K/Q rows fetched as 16B vectors from global.
// ---------------------------------------------------------------------------
__global__ __launch_bounds__(256) void k_oca(const __bf16* __restrict__ xln,
                                             const __bf16* __restrict__ xrn,
                                             const float*  __restrict__ bias,
                                             float* __restrict__ out_l,
                                             float* __restrict__ out_r) {
    __shared__ __bf16 KVt[DIMC * NKP];        // keys transposed [c][p], 30 KB
    __shared__ float  Lg[8][16 * NK];         // per-wave logits, 72 KB
    __shared__ __bf16 Pw[8][16 * NKP];        // per-wave probs (padded), 40 KB

    int blk = blockIdx.x, dir = blk & 1, wi = blk >> 1;
    int wy = wi >> 5, wx = wi & 31;
    const __bf16* Qg = dir ? xrn : xln;
    const __bf16* Kg = dir ? xln : xrn;
    float* outp = dir ? out_r : out_l;

    // stage keys transposed into KVt[c][p] (zero-pad OOB and p in [144,160))
    for (int i = threadIdx.x; i < NK * 12; i += 256) {
        int p = i / 12, ch = i % 12;
        int hh = wy * WS - 2 + p / OWIN, ww = wx * WS - 2 + p % OWIN;
        v8bf v = zero8();
        if (hh >= 0 && hh < NSIDE && ww >= 0 && ww < NSIDE)
            v = *(const v8bf*)(Kg + ((size_t)hh * NSIDE + ww) * DIMC + ch * 8);
#pragma unroll
        for (int j = 0; j < 8; ++j) KVt[(ch * 8 + j) * NKP + p] = v[j];
    }
    for (int i = threadIdx.x; i < DIMC * 16; i += 256)
        KVt[(i >> 4) * NKP + NK + (i & 15)] = (__bf16)0.f;
    __syncthreads();

    int wave = threadIdx.x >> 5, lane = threadIdx.x & 31;
    int m16 = lane & 15, hi = lane >> 4;
    float*  lg = Lg[wave];
    __bf16* pb = Pw[wave];

    for (int item = wave; item < HEADS * 4; item += 8) {
        int head = item >> 2, mt = item & 3;
        int cq = head * HD;
        // ---- B operand: query rows (scaled 0.25); channels 16..31 are zero
        int q = mt * 16 + m16;
        int th = wy * WS + (q >> 3), tw = wx * WS + (q & 7);
        const __bf16* qrow = Qg + ((size_t)th * NSIDE + tw) * DIMC + cq;
        v16bf bq;
        {
            v8bf b0 = *(const v8bf*)(qrow + hi * 8);
#pragma unroll
            for (int j = 0; j < 8; ++j) {
                bq[j]     = (__bf16)((float)b0[j] * 0.25f);
                bq[j + 8] = (__bf16)0.f;
            }
        }
        // ---- logits^T = K Q^T + bias^T : D[p_local][q_local]
        for (int nt = 0; nt < 9; ++nt) {
            int p = nt * 16 + m16;
            int hh = wy * WS - 2 + p / OWIN, ww = wx * WS - 2 + p % OWIN;
            v8bf alo = zero8();
            if (hh >= 0 && hh < NSIDE && ww >= 0 && ww < NSIDE)
                alo = *(const v8bf*)(Kg + ((size_t)hh * NSIDE + ww) * DIMC + cq + hi * 8);
            v16bf a = mk16(alo, zero8());
            v8f acc = {};
            acc = wmma_bf16(a, bq, acc);
            // lane holds q = m16, p = nt*16 + hi*8 + r (8 consecutive)
            const float* brow = bias + ((size_t)head * NQ + mt * 16 + m16) * NK
                                     + nt * 16 + hi * 8;
            float* lrow = lg + m16 * NK + nt * 16 + hi * 8;
#pragma unroll
            for (int r = 0; r < 8; ++r) lrow[r] = acc[r] + brow[r];
        }
        // ---- softmax over 144 keys (2 lanes per row) ----
        {
            int p0 = hi * 72;
            float mx = -3.4e38f;
            for (int p = 0; p < 72; ++p) mx = fmaxf(mx, lg[m16 * NK + p0 + p]);
            mx = fmaxf(mx, __shfl_xor(mx, 16, 32));
            float sm = 0.f;
            for (int p = 0; p < 72; ++p) sm += __expf(lg[m16 * NK + p0 + p] - mx);
            sm += __shfl_xor(sm, 16, 32);
            float inv = 1.f / sm;
            for (int p = 0; p < 80; ++p) {
                int pp = hi * 80 + p;
                float pr = (pp < NK) ? __expf(lg[m16 * NK + pp] - mx) * inv : 0.f;
                pb[m16 * NKP + pp] = (__bf16)pr;
            }
        }
        // ---- out = P * K  (contraction padded to 160: 5 WMMA) ----
        v8f acc2 = {};
        for (int ch = 0; ch < 5; ++ch) {
            v16bf a2 = frag_a(pb + m16 * NKP + ch * 32, hi);
            v16bf b2 = frag_b(KVt + (size_t)(cq + m16) * NKP + ch * 32, hi);
            acc2 = wmma_bf16(a2, b2, acc2);
        }
#pragma unroll
        for (int r = 0; r < 8; ++r) {
            int qq = mt * 16 + r + hi * 8;
            int oh = wy * WS + (qq >> 3), ow = wx * WS + (qq & 7);
            outp[((size_t)oh * NSIDE + ow) * DIMC + cq + m16] = acc2[r];
        }
    }
}

// ---------------------------------------------------------------------------
// K3: (optional LN) + conv1x1 96x96 via WMMA. doT transposes spatial source.
// out[u][c] = bias[c] + sum_o W[c][o] * LN(src[tok(u)])[o]
// B operand is W rows themselves ([c_out][o], contiguous over k=o).
// ---------------------------------------------------------------------------
__global__ __launch_bounds__(256) void k_conv(const float* __restrict__ src,
                                              const float* __restrict__ W,
                                              const float* __restrict__ cb,
                                              const float* __restrict__ lnw,
                                              const float* __restrict__ lnb,
                                              int doLN, int doT,
                                              __bf16* __restrict__ dst) {
    __shared__ __bf16 Wls[DIMC * DIMC];       // 18 KB
    __shared__ __bf16 As[8][16 * DIMC];       // 24 KB
    for (int i = threadIdx.x; i < DIMC * DIMC / 4; i += 256) {
        float4 wv = ((const float4*)W)[i];
        __bf16* d = Wls + i * 4;
        d[0] = (__bf16)wv.x; d[1] = (__bf16)wv.y;
        d[2] = (__bf16)wv.z; d[3] = (__bf16)wv.w;
    }
    int wave = threadIdx.x >> 5, lane = threadIdx.x & 31;
    int u0 = blockIdx.x * 128 + wave * 16;
    for (int rr = 0; rr < 16; ++rr) {
        int u = u0 + rr;
        int t = doT ? ((u & 255) * NSIDE + (u >> 8)) : u;
        const float* row = src + (size_t)t * DIMC;
        float v0 = row[lane], v1 = row[lane + 32], v2 = row[lane + 64];
        if (doLN) {
            float s = v0 + v1 + v2, s2 = v0 * v0 + v1 * v1 + v2 * v2;
            for (int off = 16; off; off >>= 1) {
                s += __shfl_xor(s, off, 32); s2 += __shfl_xor(s2, off, 32);
            }
            float mu = s * (1.f / 96.f);
            float var = s2 * (1.f / 96.f) - mu * mu;
            float r = rsqrtf(var + 1e-5f);
            v0 = (v0 - mu) * r * lnw[lane]      + lnb[lane];
            v1 = (v1 - mu) * r * lnw[lane + 32] + lnb[lane + 32];
            v2 = (v2 - mu) * r * lnw[lane + 64] + lnb[lane + 64];
        }
        As[wave][rr * DIMC + lane]      = (__bf16)v0;
        As[wave][rr * DIMC + lane + 32] = (__bf16)v1;
        As[wave][rr * DIMC + lane + 64] = (__bf16)v2;
    }
    __syncthreads();
    int m16 = lane & 15, hi = lane >> 4;
    for (int ct = 0; ct < 6; ++ct) {
        v8f acc;
        float bv = cb[ct * 16 + m16];
#pragma unroll
        for (int r = 0; r < 8; ++r) acc[r] = bv;
        for (int ch = 0; ch < 3; ++ch) {
            v16bf a = frag_a(As[wave] + m16 * DIMC + ch * 32, hi);
            v16bf b = frag_b(Wls + (ct * 16 + m16) * DIMC + ch * 32, hi);
            acc = wmma_bf16(a, b, acc);
        }
#pragma unroll
        for (int r = 0; r < 8; ++r) {
            size_t u = (size_t)u0 + r + hi * 8;
            dst[u * DIMC + ct * 16 + m16] = (__bf16)acc[r];
        }
    }
}

// ---------------------------------------------------------------------------
// K4: stage-2 logits  attn[s][i][j] = 0.25 * <Ql[s,i,:], Qr[s,j,:]>
// ---------------------------------------------------------------------------
__global__ __launch_bounds__(256) void k_qk(const __bf16* __restrict__ Ql,
                                            const __bf16* __restrict__ Qr,
                                            float* __restrict__ attn) {
    __shared__ __bf16 Qls[64 * DIMC];         // 12 KB
    __shared__ __bf16 Qrs[NSIDE * DIMC];      // 48 KB
    int s = blockIdx.x >> 2, ib = blockIdx.x & 3;
    size_t base = (size_t)s * NSIDE;
    const uint4* qsrc = (const uint4*)(Ql + (base + ib * 64) * DIMC);
    for (int i = threadIdx.x; i < 64 * DIMC / 8; i += 256) ((uint4*)Qls)[i] = qsrc[i];
    const uint4* rsrc = (const uint4*)(Qr + base * DIMC);
    for (int i = threadIdx.x; i < NSIDE * DIMC / 8; i += 256) ((uint4*)Qrs)[i] = rsrc[i];
    __syncthreads();
    int wave = threadIdx.x >> 5, lane = threadIdx.x & 31;
    int m16 = lane & 15, hi = lane >> 4;
    for (int tile = wave; tile < 64; tile += 8) {
        int it = tile >> 4, jt = tile & 15;
        v8f acc = {};
        for (int ch = 0; ch < 3; ++ch) {
            v16bf a = frag_a(Qls + (it * 16 + m16) * DIMC + ch * 32, hi);
            v16bf b = frag_b(Qrs + (jt * 16 + m16) * DIMC + ch * 32, hi);
            acc = wmma_bf16(a, b, acc);
        }
#pragma unroll
        for (int r = 0; r < 8; ++r) {
            int i = ib * 64 + it * 16 + r + hi * 8;
            int j = jt * 16 + m16;
            attn[((size_t)s * NSIDE + i) * NSIDE + j] = acc[r] * 0.25f;
        }
    }
}

// ---------------------------------------------------------------------------
// K5: softmax (rows T=0 / cols T=1) + P*V + residual + per-channel scale.
// V staged transposed [c][t] so WMMA B loads are contiguous.
// ---------------------------------------------------------------------------
__global__ __launch_bounds__(256) void k_av(const float* __restrict__ attn,
                                            const __bf16* __restrict__ V,
                                            const float* __restrict__ res,
                                            const float* __restrict__ sc,
                                            int T, float* __restrict__ out) {
    __shared__ __bf16 Pb[64 * NSIDE];         // 32 KB
    __shared__ __bf16 Vst[DIMC * NSIDE];      // V transposed [c][t], 48 KB
    __shared__ float  red[4][64];
    int s = blockIdx.x >> 2, rb = blockIdx.x & 3;
    const float* A0 = attn + (size_t)s * NSIDE * NSIDE;

    // stage V transposed
    for (int i = threadIdx.x; i < NSIDE * 12; i += 256) {
        int t = i / 12, ch = i % 12;
        v8bf v = *(const v8bf*)(V + ((size_t)s * NSIDE + t) * DIMC + ch * 8);
#pragma unroll
        for (int j = 0; j < 8; ++j) Vst[(ch * 8 + j) * NSIDE + t] = v[j];
    }

    if (T == 0) {
        int rr = threadIdx.x >> 2;        // output row (i)
        int qd = threadIdx.x & 3;
        const float* arow = A0 + ((size_t)rb * 64 + rr) * NSIDE;
        float mx = -3.4e38f;
        for (int j = qd; j < NSIDE; j += 4) mx = fmaxf(mx, arow[j]);
        mx = fmaxf(mx, __shfl_xor(mx, 1, 32));
        mx = fmaxf(mx, __shfl_xor(mx, 2, 32));
        float sm = 0.f;
        for (int j = qd; j < NSIDE; j += 4) sm += __expf(arow[j] - mx);
        sm += __shfl_xor(sm, 1, 32);
        sm += __shfl_xor(sm, 2, 32);
        float inv = 1.f / sm;
        for (int j = qd; j < NSIDE; j += 4)
            Pb[rr * NSIDE + j] = (__bf16)(__expf(arow[j] - mx) * inv);
    } else {
        int rr = threadIdx.x & 63;        // output row (j = column of attn)
        int qd = threadIdx.x >> 6;        // i quarter
        int j = rb * 64 + rr;
        float mx = -3.4e38f;
        for (int ii = qd * 64; ii < qd * 64 + 64; ++ii)
            mx = fmaxf(mx, A0[(size_t)ii * NSIDE + j]);
        red[qd][rr] = mx;
        __syncthreads();
        mx = fmaxf(fmaxf(red[0][rr], red[1][rr]), fmaxf(red[2][rr], red[3][rr]));
        __syncthreads();
        float sm = 0.f;
        for (int ii = qd * 64; ii < qd * 64 + 64; ++ii)
            sm += __expf(A0[(size_t)ii * NSIDE + j] - mx);
        red[qd][rr] = sm;
        __syncthreads();
        sm = red[0][rr] + red[1][rr] + red[2][rr] + red[3][rr];
        float inv = 1.f / sm;
        for (int ii = qd * 64; ii < qd * 64 + 64; ++ii)
            Pb[rr * NSIDE + ii] = (__bf16)(__expf(A0[(size_t)ii * NSIDE + j] - mx) * inv);
    }
    __syncthreads();

    int wave = threadIdx.x >> 5, lane = threadIdx.x & 31;
    int m16 = lane & 15, hi = lane >> 4;
    for (int tile = wave; tile < 24; tile += 8) {
        int mt = tile / 6, ct = tile % 6;
        v8f acc = {};
        for (int ch = 0; ch < 8; ++ch) {
            v16bf a = frag_a(Pb + (mt * 16 + m16) * NSIDE + ch * 32, hi);
            v16bf b = frag_b(Vst + (size_t)(ct * 16 + m16) * NSIDE + ch * 32, hi);
            acc = wmma_bf16(a, b, acc);
        }
#pragma unroll
        for (int r = 0; r < 8; ++r) {
            int u = rb * 64 + mt * 16 + r + hi * 8;
            size_t t = (size_t)s * NSIDE + u;
            int c = ct * 16 + m16;
            out[t * DIMC + c] = res[t * DIMC + c] + sc[c] * acc[r];
        }
    }
}

// ---------------------------------------------------------------------------
extern "C" void kernel_launch(void* const* d_in, const int* in_sizes, int n_in,
                              void* d_out, int out_size, void* d_ws, size_t ws_size,
                              hipStream_t stream) {
    (void)in_sizes; (void)n_in; (void)out_size; (void)ws_size;
    const float* x_l  = (const float*)d_in[0];
    const float* x_r  = (const float*)d_in[1];
    const float* rpb  = (const float*)d_in[2];
    const float* nlw  = (const float*)d_in[3];
    const float* nlb  = (const float*)d_in[4];
    const float* nrw  = (const float*)d_in[5];
    const float* nrb  = (const float*)d_in[6];
    const float* pl1w = (const float*)d_in[7];
    const float* pl1b = (const float*)d_in[8];
    const float* pr1w = (const float*)d_in[9];
    const float* pr1b = (const float*)d_in[10];
    const float* pl2w = (const float*)d_in[11];
    const float* pl2b = (const float*)d_in[12];
    const float* pr2w = (const float*)d_in[13];
    const float* pr2b = (const float*)d_in[14];
    const float* beta  = (const float*)d_in[15];
    const float* gamma = (const float*)d_in[16];

    char* ws = (char*)d_ws;
    const size_t SZ_BIAS = (size_t)HEADS * NQ * NK * sizeof(float);
    const size_t SZ_BF   = (size_t)NTOK * DIMC * sizeof(__bf16);
    const size_t SZ_F32  = (size_t)NTOK * DIMC * sizeof(float);
    size_t off = 0;
    float*  bias = (float*) (ws + off); off += (SZ_BIAS + 255) & ~(size_t)255;
    __bf16* xln  = (__bf16*)(ws + off); off += SZ_BF;
    __bf16* xrn  = (__bf16*)(ws + off); off += SZ_BF;
    float*  xl1  = (float*) (ws + off); off += SZ_F32;
    float*  xr1  = (float*) (ws + off); off += SZ_F32;
    __bf16* Qlb  = (__bf16*)(ws + off); off += SZ_BF;
    __bf16* Qrb  = (__bf16*)(ws + off); off += SZ_BF;
    __bf16* Vlb  = (__bf16*)(ws + off); off += SZ_BF;
    __bf16* Vrb  = (__bf16*)(ws + off); off += SZ_BF;
    float*  attn = (float*) (ws + off); off += (size_t)NSIDE * NSIDE * NSIDE * sizeof(float);

    float* out_l = (float*)d_out;
    float* out_r = out_l + (size_t)NTOK * DIMC;

    k_bias<<<(HEADS * NQ * NK + 255) / 256, 256, 0, stream>>>(rpb, bias);
    k_ln<<<NTOK / 8, 256, 0, stream>>>(x_l, nlw, nlb, xln);
    k_ln<<<NTOK / 8, 256, 0, stream>>>(x_r, nlw, nlb, xrn);

    k_oca<<<2048, 256, 0, stream>>>(xln, xrn, bias, xl1, xr1);

    k_conv<<<NTOK / 128, 256, 0, stream>>>(xl1, pl1w, pl1b, nlw, nlb, 1, 1, Qlb);
    k_conv<<<NTOK / 128, 256, 0, stream>>>(xr1, pr1w, pr1b, nrw, nrb, 1, 1, Qrb);
    k_conv<<<NTOK / 128, 256, 0, stream>>>(x_l, pl2w, pl2b, nlw, nlb, 0, 0, Vlb);
    k_conv<<<NTOK / 128, 256, 0, stream>>>(x_r, pr2w, pr2b, nlw, nlb, 0, 0, Vrb);

    k_qk<<<NSIDE * 4, 256, 0, stream>>>(Qlb, Qrb, attn);
    k_av<<<NSIDE * 4, 256, 0, stream>>>(attn, Vrb, x_l, beta,  0, out_l);
    k_av<<<NSIDE * 4, 256, 0, stream>>>(attn, Vlb, x_r, gamma, 1, out_r);
}
